// CapsNet_6511170421229
// MI455X (gfx1250) — matrix operand order repaired and checked by reading, not compiled, not run on hardware
//
#include <hip/hip_runtime.h>
#include <hip/hip_bf16.h>
#include <math.h>

typedef __bf16 bf16;
typedef __attribute__((ext_vector_type(16))) __bf16 bf16x16;
typedef __attribute__((ext_vector_type(8)))  float  f32x8;

#define BATCH 128
#define SEQT  800
#define EDIM  100
#define EPAD  128
#define HDIM  96
#define D2    192
#define GN    384          // padded gate columns: [r|z|gi_n|gh_n]
#define AK    224          // fused K: 128 (x, padded) + 96 (h)
#define KT7   7
#define NT24  24
#define RCAPS 12
#define CCAPS 50
#define ODIM  16
#define NTROWS (BATCH*SEQT)   // 102400
#define WFRAG_EL (KT7*NT24*32*16)  // 86016 per direction

// ---------------- WMMA helpers (A/B k order: contiguous k = half*16 + s) ----------------

__device__ __forceinline__ f32x8 wmma_bf16(bf16x16 a, bf16x16 b, f32x8 c) {
  return __builtin_amdgcn_wmma_f32_16x16x32_bf16(false, a, false, b, (short)0, c,
                                                 false, false);
}

union BF16Frag { bf16x16 v; uint4 q[2]; };

// A-matrix 16x32 bf16: lane m=lane&15; elems 0..15 = K half*16+0..15
__device__ __forceinline__ bf16x16 load_afrag(const bf16* base, int rowstride,
                                              int kofs, int lane) {
  const bf16* p = base + (size_t)(lane & 15) * rowstride + kofs + ((lane >> 4) << 4);
  BF16Frag f;
  f.q[0] = *(const uint4*)p;
  f.q[1] = *(const uint4*)(p + 8);
  return f.v;
}

// B-matrix 32x16 bf16, pre-arranged so each lane's 16 halves are contiguous
__device__ __forceinline__ bf16x16 load_bfrag16(const bf16* p) {
  BF16Frag f;
  f.q[0] = *(const uint4*)p;
  f.q[1] = *(const uint4*)(p + 8);
  return f.v;
}

__device__ __forceinline__ float sigmoidf_(float x) { return 1.f / (1.f + __expf(-x)); }

// ---------------- Prep kernels: bf16 + fragment-order weight swizzles --------

// Fused GRU weight matrix [AK=224 x GN=384] per direction, stored in B-fragment
// order [kt][nt][lane][16]:  K = kt*32 + half*16 + p,  N = nt*16 + (lane&15)
__global__ void k_prep_wfrag(const float* __restrict__ wih_f, const float* __restrict__ whh_f,
                             const float* __restrict__ wih_b, const float* __restrict__ whh_b,
                             bf16* __restrict__ wfrag) {
  int idx = blockIdx.x * blockDim.x + threadIdx.x;   // 2*WFRAG_EL
  int dir = idx / WFRAG_EL, r = idx % WFRAG_EL;
  const float* wih = dir ? wih_b : wih_f;            // [288][100]
  const float* whh = dir ? whh_b : whh_f;            // [288][96]
  int kt = r / (NT24 * 512); int r2 = r % (NT24 * 512);
  int nt = r2 / 512;         int r3 = r2 % 512;
  int lane = r3 >> 4, p = r3 & 15;
  int K = kt * 32 + ((lane >> 4) << 4) + p;
  int N = nt * 16 + (lane & 15);
  float v = 0.f;
  if (N < 192) {                               // r,z gates: x-part + h-part summed
    if (K < EPAD) { if (K < EDIM) v = wih[N * EDIM + K]; }
    else          { int kk = K - EPAD; if (kk < HDIM) v = whh[N * HDIM + kk]; }
  } else if (N < 288) {                        // gi_n (x-part of n gate)
    if (K < EDIM) v = wih[N * EDIM + K];
  } else {                                     // gh_n (h-part of n gate)
    if (K >= EPAD) { int kk = K - EPAD; if (kk < HDIM) v = whh[(N - 96) * HDIM + kk]; }
  }
  wfrag[idx] = (bf16)v;
}

// combined biases [2][384], A1 fragments [6][4][32][16], zero the att_reg accum
__global__ void k_prep_misc(const float* __restrict__ bih_f, const float* __restrict__ bhh_f,
                            const float* __restrict__ bih_b, const float* __restrict__ bhh_b,
                            const float* __restrict__ A1, float* __restrict__ bcomb,
                            bf16* __restrict__ a1frag, float* __restrict__ accum) {
  int idx = blockIdx.x * blockDim.x + threadIdx.x;
  if (idx == 0) *accum = 0.f;
  if (idx < 2 * GN) {
    int dir = idx / GN, c = idx % GN;
    const float* bih = dir ? bih_b : bih_f;
    const float* bhh = dir ? bhh_b : bhh_f;
    float v;
    if (c < 192) v = bih[c] + bhh[c];
    else if (c < 288) v = bih[c];
    else v = bhh[c - 96];
    bcomb[idx] = v;
  }
  int e = idx - 2 * GN;
  if (e >= 0 && e < 6 * 4 * 512) {
    int kt = e / 2048, r2 = e % 2048;
    int nt = r2 / 512, r3 = r2 % 512;
    int lane = r3 >> 4, p = r3 & 15;
    int K = kt * 32 + ((lane >> 4) << 4) + p;
    int N = nt * 16 + (lane & 15);
    a1frag[e] = (bf16)((N < 50) ? A1[K * 50 + N] : 0.f);
  }
}

__global__ void k_prep_wcaps(const float* __restrict__ Wcaps, bf16* __restrict__ wcapsbf) {
  int idx = blockIdx.x * blockDim.x + threadIdx.x;  // 12*192*800
  wcapsbf[idx] = (bf16)Wcaps[idx];
}

// ---------------- Embedding gather (padded to 128 cols, bf16) ---------------

__global__ void k_embed(const int* __restrict__ tokens, const float* __restrict__ emb,
                        bf16* __restrict__ xxbf) {
  int idx = blockIdx.x * blockDim.x + threadIdx.x;  // NTROWS*EPAD
  int row = idx >> 7, e = idx & 127;
  int tok = tokens[row];
  float v = (e < EDIM) ? emb[(size_t)tok * EDIM + e] : 0.f;
  xxbf[idx] = (bf16)v;
}

// ---------------- Fused bidirectional GRU (persistent per-row-slab) ---------
// grid (50, 2): 16 rows per WG, dir = blockIdx.y. 256 thr = 8 waves, 3 N-tiles
// per wave. Weights (224x384 bf16, fragment order) resident in LDS all 128 steps.

__global__ void k_gru(const bf16* __restrict__ xxbf, const bf16* __restrict__ wfrag,
                      const float* __restrict__ bcomb, bf16* __restrict__ hbf,
                      bf16* __restrict__ hbfT) {
  extern __shared__ char smem[];
  bf16*  wlds   = (bf16*)smem;                  // 86016 el (172032 B)
  bf16*  astage = wlds + WFRAG_EL;              // 16*224 el (7168 B)
  float* gbuf   = (float*)(astage + 16 * AK);   // 16*384 f32 (24576 B)
  float* hbuf   = gbuf + 16 * GN;               // 16*96  f32 (6144 B)
  float* bl     = hbuf + 16 * HDIM;             // 384 f32   (1536 B)

  const int dir = blockIdx.y;
  const int n0  = blockIdx.x * 16;
  const int tid = threadIdx.x;
  const int lane = tid & 31;
  const int wave = tid >> 5;

  { // stage fused weights (fragment order -> linear copy)
    const uint4* src = (const uint4*)(wfrag + (size_t)dir * WFRAG_EL);
    uint4* dst = (uint4*)wlds;
    for (int i = tid; i < WFRAG_EL / 8; i += 256) dst[i] = src[i];
  }
  for (int i = tid; i < GN; i += 256) bl[i] = bcomb[dir * GN + i];
  for (int i = tid; i < 16 * HDIM; i += 256) hbuf[i] = 0.f;
  for (int i = tid; i < 16 * AK; i += 256) astage[i] = (bf16)0.f;
  __syncthreads();

  for (int s = 0; s < BATCH; ++s) {
    const int seff = dir ? (BATCH - 1 - s) : s;
    { // stage x rows: 16 rows x 256B, one uint4 per thread
      int m = tid >> 4, ch = tid & 15;
      const uint4* src = (const uint4*)(xxbf + ((size_t)seff * SEQT + n0 + m) * EPAD + ch * 8);
      *(uint4*)(astage + m * AK + ch * 8) = *src;
      if (s + 1 < BATCH) {
        int sn = dir ? (BATCH - 2 - s) : (s + 1);
        __builtin_prefetch(xxbf + ((size_t)sn * SEQT + n0 + m) * EPAD, 0, 1);
      }
    }
    __syncthreads();

    f32x8 zero = {0.f,0.f,0.f,0.f,0.f,0.f,0.f,0.f};
    f32x8 acc[3] = {zero, zero, zero};
    #pragma unroll
    for (int kt = 0; kt < KT7; ++kt) {
      bf16x16 af = load_afrag(astage, AK, kt * 32, lane);
      #pragma unroll
      for (int j = 0; j < 3; ++j) {
        int nt = wave * 3 + j;
        bf16x16 bfv = load_bfrag16(wlds + (((kt * NT24 + nt) * 32 + lane) << 4));
        acc[j] = wmma_bf16(af, bfv, acc[j]);
      }
    }
    #pragma unroll
    for (int j = 0; j < 3; ++j) {
      int nt = wave * 3 + j;
      #pragma unroll
      for (int v = 0; v < 8; ++v)
        gbuf[(((lane >> 4) << 3) + v) * GN + nt * 16 + (lane & 15)] = acc[j][v];
    }
    __syncthreads();

    // gate math on 16*96 h elements
    for (int e = tid; e < 16 * HDIM; e += 256) {
      int m = e / HDIM, c = e - m * HDIM;
      const float* g = gbuf + m * GN;
      float r  = sigmoidf_(g[c]        + bl[c]);
      float z  = sigmoidf_(g[96 + c]   + bl[96 + c]);
      float nn = tanhf((g[192 + c] + bl[192 + c]) + r * (g[288 + c] + bl[288 + c]));
      float hold = hbuf[e];
      float hnew = (1.f - z) * nn + z * hold;
      hbuf[e] = hnew;
      astage[m * AK + EPAD + c] = (bf16)hnew;             // feed next step's A
      size_t grow = (size_t)seff * SEQT + n0 + m;
      hbf[grow * D2 + dir * HDIM + c] = (bf16)hnew;       // [bt][192]
      hbfT[((size_t)seff * D2 + dir * HDIM + c) * SEQT + (n0 + m)] = (bf16)hnew; // [b][192][t]
    }
    __syncthreads();
  }
}

// ---------------- Attention: tanh(h@A1+b1)@A2+b2, softmax over R -------------
// grid 1600 x 64 rows. WMMA for h@A1 (K=192), VALU for the tiny 50->12 stage.

__global__ void k_attn(const bf16* __restrict__ hbf, const bf16* __restrict__ a1frag,
                       const float* __restrict__ b1, const float* __restrict__ A2,
                       const float* __restrict__ b2, float* __restrict__ att,
                       bf16* __restrict__ attbf) {
  extern __shared__ char smem[];
  bf16*  a1l = (bf16*)smem;                 // 12288 el (24576 B)
  float* Pls = (float*)(smem + 24576);      // 64*64 f32
  float* Sls = Pls + 64 * 64;               // 64*12 f32

  const int tid = threadIdx.x, lane = tid & 31, wave = tid >> 5;
  const int row0 = blockIdx.x * 64;

  for (int i = tid; i < 12288 / 8; i += 256) ((uint4*)a1l)[i] = ((const uint4*)a1frag)[i];
  __syncthreads();

  const int mt = wave >> 1;
  f32x8 zero = {0.f,0.f,0.f,0.f,0.f,0.f,0.f,0.f};
  f32x8 acc[2] = {zero, zero};
  #pragma unroll
  for (int kt = 0; kt < 6; ++kt) {
    bf16x16 af = load_afrag(hbf + ((size_t)(row0 + mt * 16)) * D2 + kt * 32, D2, 0, lane);
    #pragma unroll
    for (int j = 0; j < 2; ++j) {
      int nt = (wave & 1) * 2 + j;
      bf16x16 bfv = load_bfrag16(a1l + (((kt * 4 + nt) * 32 + lane) << 4));
      acc[j] = wmma_bf16(af, bfv, acc[j]);
    }
  }
  #pragma unroll
  for (int j = 0; j < 2; ++j) {
    int nt = (wave & 1) * 2 + j;
    #pragma unroll
    for (int v = 0; v < 8; ++v) {
      int rr = mt * 16 + ((lane >> 4) << 3) + v;
      int cc = nt * 16 + (lane & 15);
      Pls[rr * 64 + cc] = (cc < 50) ? tanhf(acc[j][v] + b1[cc]) : 0.f;
    }
  }
  __syncthreads();
  for (int e = tid; e < 64 * 12; e += 256) {
    int rr = e / 12, r = e - rr * 12;
    float sacc = b2[r];
    for (int jj = 0; jj < 50; ++jj) sacc += Pls[rr * 64 + jj] * A2[jj * 12 + r];
    Sls[e] = sacc;
  }
  __syncthreads();
  if (tid < 64) {
    float mx = -1e30f;
    for (int r = 0; r < 12; ++r) mx = fmaxf(mx, Sls[tid * 12 + r]);
    float ex[12], sum = 0.f;
    for (int r = 0; r < 12; ++r) { ex[r] = __expf(Sls[tid * 12 + r] - mx); sum += ex[r]; }
    float inv = 1.f / sum;
    size_t gr = (size_t)row0 + tid;
    for (int r = 0; r < 12; ++r) {
      float a = ex[r] * inv;
      att[gr * 12 + r] = a;
      attbf[gr * 16 + r] = (bf16)a;
    }
    for (int r = 12; r < 16; ++r) attbf[gr * 16 + r] = (bf16)0.f;
  }
}

// ---------------- m = att^T @ h per batch (K=800) + attention regularizer ----
// grid 128, 192 thr = 6 waves x 2 M-tiles (h dim 192), N = one 16-col tile (R=12).

__global__ void k_mg(const bf16* __restrict__ hbfT, const bf16* __restrict__ attbf,
                     const float* __restrict__ att, float* __restrict__ mbuf,
                     float* __restrict__ accum) {
  extern __shared__ char smem[];
  bf16*  attlsT = (bf16*)smem;              // 16 cols x 32 k
  float* gls    = (float*)(smem + 1024);    // 144

  const int b = blockIdx.x;
  const int tid = threadIdx.x, lane = tid & 31, wave = tid >> 5;
  f32x8 zero = {0.f,0.f,0.f,0.f,0.f,0.f,0.f,0.f};
  f32x8 acc[2] = {zero, zero};

  for (int kt = 0; kt < 25; ++kt) {
    for (int e = tid; e < 512; e += 192) {  // transpose-stage att tile
      int n = e >> 5, k = e & 31;
      attlsT[n * 32 + k] = attbf[((size_t)b * SEQT + kt * 32 + k) * 16 + n];
    }
    __syncthreads();
    bf16x16 bfv = load_bfrag16(attlsT + (lane & 15) * 32 + ((lane >> 4) << 4));
    #pragma unroll
    for (int j = 0; j < 2; ++j) {
      int mt = wave * 2 + j;
      bf16x16 af = load_afrag(hbfT + ((size_t)b * D2 + mt * 16) * SEQT + kt * 32,
                              SEQT, 0, lane);
      acc[j] = wmma_bf16(af, bfv, acc[j]);
    }
    __syncthreads();
  }
  int col = lane & 15;
  #pragma unroll
  for (int j = 0; j < 2; ++j) {
    int mt = wave * 2 + j;
    if (col < 12) {
      #pragma unroll
      for (int v = 0; v < 8; ++v) {
        int hidx = mt * 16 + ((lane >> 4) << 3) + v;
        mbuf[((size_t)b * RCAPS + col) * D2 + hidx] = acc[j][v];
      }
    }
  }
  __syncthreads();
  if (tid < 144) {
    int r = tid / 12, s2 = tid % 12;
    const float* ab = att + (size_t)b * SEQT * 12;
    float a2 = 0.f;
    for (int t = 0; t < SEQT; ++t) a2 += ab[t * 12 + r] * ab[t * 12 + s2];
    if (r == s2) a2 -= 1.f;
    gls[tid] = a2 * a2;
  }
  __syncthreads();
  if (tid == 0) {
    float sum = 0.f;
    for (int i = 0; i < 144; ++i) sum += gls[i];
    atomicAdd(accum, sqrtf(sum) * (1.f / 128.f));
  }
}

// ---------------- squash(m) -> v0 (bf16), emit att_reg -----------------------

__global__ void k_squash(const float* __restrict__ mbuf, bf16* __restrict__ v0bf,
                         const float* __restrict__ accum, float* __restrict__ dout) {
  int idx = blockIdx.x * blockDim.x + threadIdx.x;
  if (idx == 0) dout[NTROWS] = *accum;
  if (idx >= BATCH * RCAPS) return;
  const float* x = mbuf + (size_t)idx * D2;
  float n = 0.f;
  for (int i = 0; i < D2; ++i) n += x[i] * x[i];
  float sc = n / ((n + 1.f) * sqrtf(n + 1e-30f));
  bf16* o = v0bf + (size_t)idx * D2;
  for (int i = 0; i < D2; ++i) o[i] = (bf16)(x[i] * sc);
}

// ---------------- u_hat = v0 @ Wcaps per primary capsule (K=192) -------------
// grid (12, 5): capsule r, 160-col chunk of C*O=800. M=128 batches (8 M-tiles).

__global__ void k_uhat(const bf16* __restrict__ v0bf, const bf16* __restrict__ wcapsbf,
                       float* __restrict__ uhat) {
  extern __shared__ char smem[];
  bf16* Bls = (bf16*)smem;                  // 160 cols x 32 k
  const int r = blockIdx.x, n0 = blockIdx.y * 160;
  const int tid = threadIdx.x, lane = tid & 31, mt = tid >> 5;
  f32x8 zero = {0.f,0.f,0.f,0.f,0.f,0.f,0.f,0.f};
  f32x8 acc[10];
  #pragma unroll
  for (int j = 0; j < 10; ++j) acc[j] = zero;

  for (int kt = 0; kt < 6; ++kt) {
    for (int e = tid; e < 5120; e += 256) {
      int nl = e % 160, k = e / 160;
      Bls[nl * 32 + k] = wcapsbf[((size_t)r * D2 + kt * 32 + k) * 800 + n0 + nl];
    }
    __syncthreads();
    bf16x16 af = load_afrag(v0bf + ((size_t)(mt * 16) * RCAPS + r) * D2 + kt * 32,
                            RCAPS * D2, 0, lane);
    #pragma unroll
    for (int j = 0; j < 10; ++j) {
      bf16x16 bfv = load_bfrag16(Bls + (j * 16 + (lane & 15)) * 32 + ((lane >> 4) << 4));
      acc[j] = wmma_bf16(af, bfv, acc[j]);
    }
    __syncthreads();
  }
  #pragma unroll
  for (int j = 0; j < 10; ++j) {
    int cc = n0 + j * 16 + (lane & 15);
    #pragma unroll
    for (int v = 0; v < 8; ++v) {
      int bb = mt * 16 + ((lane >> 4) << 3) + v;
      uhat[((size_t)bb * RCAPS + r) * 800 + cc] = acc[j][v];
    }
  }
}

// ---------------- Dynamic routing (3 iters) in LDS, one WG per batch ---------

__global__ void k_route(const float* __restrict__ uhat, float* __restrict__ dout) {
  extern __shared__ char smem[];
  float* uls  = (float*)smem;      // 12*800
  float* blog = uls + 9600;        // 12*50
  float* cls  = blog + 600;        // 12*50
  float* sls  = cls + 600;         // 50*16
  float* vls  = sls + 800;         // 50*16
  const int b = blockIdx.x, tid = threadIdx.x;

  for (int e = tid; e < 9600; e += 256) uls[e] = uhat[(size_t)b * 9600 + e];
  for (int e = tid; e < 600; e += 256) blog[e] = 0.f;
  __syncthreads();

  for (int it = 0; it < 3; ++it) {
    if (tid < 12) {
      float mx = -1e30f;
      for (int c = 0; c < 50; ++c) mx = fmaxf(mx, blog[tid * 50 + c]);
      float sum = 0.f;
      for (int c = 0; c < 50; ++c) sum += __expf(blog[tid * 50 + c] - mx);
      float inv = 1.f / sum;
      for (int c = 0; c < 50; ++c) cls[tid * 50 + c] = __expf(blog[tid * 50 + c] - mx) * inv;
    }
    __syncthreads();
    for (int e = tid; e < 800; e += 256) {
      int c = e >> 4;
      float sacc = 0.f;
      for (int rr = 0; rr < 12; ++rr) sacc += cls[rr * 50 + c] * uls[rr * 800 + e];
      sls[e] = sacc;
    }
    __syncthreads();
    if (tid < 50) {
      float n = 0.f;
      for (int o = 0; o < 16; ++o) { float xv = sls[tid * 16 + o]; n += xv * xv; }
      float sc = n / ((n + 1.f) * sqrtf(n + 1e-30f));
      for (int o = 0; o < 16; ++o) vls[tid * 16 + o] = sls[tid * 16 + o] * sc;
    }
    __syncthreads();
    if (it < 2) {
      for (int e = tid; e < 600; e += 256) {
        int rr = e / 50, c = e % 50;
        float d = 0.f;
        for (int o = 0; o < 16; ++o) d += uls[rr * 800 + c * 16 + o] * vls[c * 16 + o];
        blog[e] += d;
      }
      __syncthreads();
    }
  }
  for (int e = tid; e < 800; e += 256) dout[(size_t)b * 800 + e] = vls[e];
}

// ---------------- Launch ------------------------------------------------------

extern "C" void kernel_launch(void* const* d_in, const int* in_sizes, int n_in,
                              void* d_out, int out_size, void* d_ws, size_t ws_size,
                              hipStream_t stream) {
  (void)in_sizes; (void)n_in; (void)out_size; (void)ws_size;
  const int*   tokens = (const int*)d_in[0];
  const float* emb    = (const float*)d_in[1];
  const float* wih_f  = (const float*)d_in[2];
  const float* whh_f  = (const float*)d_in[3];
  const float* bih_f  = (const float*)d_in[4];
  const float* bhh_f  = (const float*)d_in[5];
  const float* wih_b  = (const float*)d_in[6];
  const float* whh_b  = (const float*)d_in[7];
  const float* bih_b  = (const float*)d_in[8];
  const float* bhh_b  = (const float*)d_in[9];
  const float* A1     = (const float*)d_in[10];
  const float* b1     = (const float*)d_in[11];
  const float* A2     = (const float*)d_in[12];
  const float* b2     = (const float*)d_in[13];
  const float* Wcaps  = (const float*)d_in[14];
  float* dout = (float*)d_out;

  char* w = (char*)d_ws;
  auto carve = [&](size_t bytes) { char* p = w; w += (bytes + 255) & ~(size_t)255; return p; };
  bf16*  xxbf    = (bf16*)carve((size_t)NTROWS * EPAD * 2);          // 26.2 MB
  bf16*  hbf     = (bf16*)carve((size_t)NTROWS * D2 * 2);            // 39.3 MB
  bf16*  hbfT    = (bf16*)carve((size_t)BATCH * D2 * SEQT * 2);      // 39.3 MB
  float* att     = (float*)carve((size_t)NTROWS * 12 * 4);           //  4.9 MB
  bf16*  attbf   = (bf16*)carve((size_t)NTROWS * 16 * 2);            //  3.3 MB
  float* mbuf    = (float*)carve((size_t)BATCH * RCAPS * D2 * 4);
  bf16*  v0bf    = (bf16*)carve((size_t)BATCH * RCAPS * D2 * 2);
  float* uhat    = (float*)carve((size_t)BATCH * RCAPS * 800 * 4);   //  4.9 MB
  bf16*  wfrag   = (bf16*)carve((size_t)2 * WFRAG_EL * 2);
  float* bcomb   = (float*)carve(2 * GN * 4);
  bf16*  a1frag  = (bf16*)carve(6 * 4 * 512 * 2);
  bf16*  wcapsbf = (bf16*)carve((size_t)RCAPS * D2 * 800 * 2);       //  3.7 MB
  float* accum   = (float*)carve(256);

  k_prep_wfrag<<<(2 * WFRAG_EL) / 256, 256, 0, stream>>>(wih_f, whh_f, wih_b, whh_b, wfrag);
  k_prep_misc <<<52, 256, 0, stream>>>(bih_f, bhh_f, bih_b, bhh_b, A1, bcomb, a1frag, accum);
  k_prep_wcaps<<<(RCAPS * D2 * 800) / 256, 256, 0, stream>>>(Wcaps, wcapsbf);
  k_embed<<<((size_t)NTROWS * EPAD) / 256, 256, 0, stream>>>(tokens, emb, xxbf);
  k_gru<<<dim3(SEQT / 16, 2), 256, 211456, stream>>>(xxbf, wfrag, bcomb, hbf, hbfT);
  k_attn<<<NTROWS / 64, 256, 44032, stream>>>(hbf, a1frag, b1, A2, b2, att, attbf);
  k_mg<<<BATCH, 192, 2048, stream>>>(hbfT, attbf, att, mbuf, accum);
  k_squash<<<6, 256, 0, stream>>>(mbuf, v0bf, accum, dout);
  k_uhat<<<dim3(RCAPS, 5), 256, 10240, stream>>>(v0bf, wcapsbf, uhat);
  k_route<<<BATCH, 256, 49600, stream>>>(uhat, dout);
}
